// GroupQuantLinear_11733850653104
// MI455X (gfx1250) — matrix operand order repaired and checked by reading, not compiled, not run on hardware
//
#include <hip/hip_runtime.h>

typedef __attribute__((ext_vector_type(16))) _Float16 v16h;
typedef __attribute__((ext_vector_type(8)))  _Float16 v8h;
typedef __attribute__((ext_vector_type(2)))  _Float16 h2;
typedef __attribute__((ext_vector_type(8)))  float    v8f;
typedef __attribute__((ext_vector_type(4)))  float    f4;
typedef __attribute__((ext_vector_type(4)))  int      i4;

// Fixed problem dims from the reference
constexpr int MDIM = 2 * 2048;   // B*S
constexpr int KDIM = 4096;       // IN
constexpr int NDIM = 16384;      // OUT
constexpr int G    = 16;
constexpr int GRP  = KDIM / G;   // 256 dequantized values per group
constexpr int WPG  = GRP / 4;    // 64 packed int32 words per group

constexpr int BM = 256, BN = 128, BK = 32;
constexpr int LDSS = BK + 8;     // 40 halves = 20 dwords row stride -> conflict-free b128 frag loads
constexpr int NK = KDIM / BK;    // 128 K-steps

__global__ __launch_bounds__(256)
void gql_wmma_kernel(const float* __restrict__ X,
                     const int*   __restrict__ Wp,
                     const float* __restrict__ Ws,
                     const float* __restrict__ Wb,
                     const float* __restrict__ Bias,
                     float* __restrict__ Y)
{
    __shared__ _Float16 lA[2][BM * LDSS];   // 2 x 20 KB
    __shared__ _Float16 lB[2][BN * LDSS];   // 2 x 10 KB

    const int tid  = threadIdx.x;
    const int lane = tid & 31;
    const int wave = tid >> 5;

    const int m0 = blockIdx.y * BM;
    const int n0 = blockIdx.x * BN;

    // wave tile 64(M) x 64(N); wave grid 4(M) x 2(N)
    const int wm = (wave >> 1) * 64;
    const int wn = (wave & 1) * 64;

    // A staging: 8 rounds; row = tid/8 + r*32, 4 floats at col (tid%8)*4
    const int a_row = tid >> 3;
    const int a_col = (tid & 7) * 4;
    // W staging: row = tid/2; half = tid&1 covers k = half*16 .. +15 (4 packed words)
    const int b_row  = tid >> 1;
    const int b_half = tid & 1;

    v8f acc[4][4];
#pragma unroll
    for (int i = 0; i < 4; ++i)
#pragma unroll
        for (int j = 0; j < 4; ++j) acc[i][j] = (v8f)0.f;

    f4 aReg[8];
    i4 wReg;
    float wsc, wbi;

    auto loadStage = [&](int kb) {
        const int kbase = kb * BK;
#pragma unroll
        for (int r = 0; r < 8; ++r) {
            const int m = m0 + a_row + r * 32;
            aReg[r] = *(const f4*)(X + (size_t)m * KDIM + kbase + a_col);
        }
        const int n     = n0 + b_row;
        const int g     = kbase / GRP;
        const int wbase = ((kbase % GRP) >> 2) + b_half * 4;
        wReg = *(const i4*)(Wp + (size_t)n * (G * WPG) + g * WPG + wbase);
        wsc  = Ws[n * G + g];
        wbi  = Wb[n * G + g];
    };

    auto storeStage = [&](int buf) {
        // X tile: f32 -> packed f16 (v_cvt_pkrtz_f16_f32), 8B stores
#pragma unroll
        for (int r = 0; r < 8; ++r) {
            uint2 u;
            u.x = __builtin_bit_cast(unsigned,
                      __builtin_amdgcn_cvt_pkrtz(aReg[r].x, aReg[r].y));
            u.y = __builtin_bit_cast(unsigned,
                      __builtin_amdgcn_cvt_pkrtz(aReg[r].z, aReg[r].w));
            *(uint2*)&lA[buf][(a_row + r * 32) * LDSS + a_col] = u;
        }
        // int4 -> f16 magic dequant: 0x6400|q == 1024+q exactly in fp16.
        // tA/tB written to fuse as v_lshl_or_b32 + v_and_or_b32.
        const h2 s2 = { (_Float16)wsc, (_Float16)wsc };
        const h2 b2 = { (_Float16)wbi, (_Float16)wbi };
        const h2 nk = { (_Float16)-1024.0f, (_Float16)-1024.0f };
        unsigned out[8];
#pragma unroll
        for (int w = 0; w < 4; ++w) {
            const unsigned word = (unsigned)wReg[w];   // nibbles q3q2q1q0 in bits[15:0]
            const unsigned u    = word >> 8;
            unsigned tA = (((word << 12) | word) & 0x000F000Fu) | 0x64006400u; // (q0,q1)
            unsigned tB = (((u    << 12) | u   ) & 0x000F000Fu) | 0x64006400u; // (q2,q3)
            h2 qA = __builtin_bit_cast(h2, tA) + nk;   // exact (q0,q1)
            h2 qB = __builtin_bit_cast(h2, tB) + nk;   // exact (q2,q3)
            h2 vA = qA * s2 + b2;                      // v_pk_fma_f16
            h2 vB = qB * s2 + b2;
            out[2 * w]     = __builtin_bit_cast(unsigned, vA);
            out[2 * w + 1] = __builtin_bit_cast(unsigned, vB);
        }
        unsigned* q = (unsigned*)&lB[buf][b_row * LDSS + b_half * 16];
        uint4 d0 = { out[0], out[1], out[2], out[3] };
        uint4 d1 = { out[4], out[5], out[6], out[7] };
        *(uint4*)(q)     = d0;
        *(uint4*)(q + 4) = d1;
    };

    const int half = lane >> 4;   // K-half selector per ISA fragment layout
    const int l16  = lane & 15;

    auto compute = [&](int buf) {
        // Load all four B fragments once; stream A fragments one at a time
        v16h bfrag[4];
#pragma unroll
        for (int ni = 0; ni < 4; ++ni) {
            const _Float16* base = &lB[buf][(wn + ni * 16 + l16) * LDSS];
            v8h lo = *(const v8h*)(base + half * 8);        // K = half*8 + 0..7
            v8h hi = *(const v8h*)(base + 16 + half * 8);   // K = 16 + half*8 + 0..7
            v16h b;
#pragma unroll
            for (int e = 0; e < 8; ++e) { b[e] = lo[e]; b[8 + e] = hi[e]; }
            bfrag[ni] = b;
        }
#pragma unroll
        for (int mi = 0; mi < 4; ++mi) {
            const _Float16* base = &lA[buf][(wm + mi * 16 + l16) * LDSS];
            v8h lo = *(const v8h*)(base + half * 8);        // K = half*8 + 0..7
            v8h hi = *(const v8h*)(base + 16 + half * 8);   // K = 16 + half*8 + 0..7
            v16h a;
#pragma unroll
            for (int e = 0; e < 8; ++e) { a[e] = lo[e]; a[8 + e] = hi[e]; }
#pragma unroll
            for (int ni = 0; ni < 4; ++ni)
                acc[mi][ni] = __builtin_amdgcn_wmma_f32_16x16x32_f16(
                    false, a, false, bfrag[ni],
                    (short)0, acc[mi][ni], false, false);
        }
    };

    // Prologue
    loadStage(0);
    storeStage(0);
    __syncthreads();

    // Pipelined main loop: fetch kb+1 while computing kb; one barrier/iter
    for (int kb = 0; kb < NK; ++kb) {
        const int cur = kb & 1;
        if (kb + 1 < NK) loadStage(kb + 1);
        compute(cur);
        if (kb + 1 < NK) storeStage(cur ^ 1);
        __syncthreads();
    }

    // Epilogue: C/D layout -> lane = N, VGPR r -> M = (lane/16)*8 + r
#pragma unroll
    for (int mi = 0; mi < 4; ++mi) {
#pragma unroll
        for (int ni = 0; ni < 4; ++ni) {
            const int   n  = n0 + wn + ni * 16 + l16;
            const float bv = Bias[n];
#pragma unroll
            for (int r = 0; r < 8; ++r) {
                const int m = m0 + wm + mi * 16 + half * 8 + r;
                __builtin_nontemporal_store(acc[mi][ni][r] + bv,
                                            Y + (size_t)m * NDIM + n);
            }
        }
    }
}

extern "C" void kernel_launch(void* const* d_in, const int* in_sizes, int n_in,
                              void* d_out, int out_size, void* d_ws, size_t ws_size,
                              hipStream_t stream) {
    const float* X    = (const float*)d_in[0];  // [B,S,IN] f32
    const int*   Wp   = (const int*)d_in[1];    // [OUT,G,64] i32
    const float* Ws   = (const float*)d_in[2];  // [OUT,G,1] f32
    const float* Wb   = (const float*)d_in[3];  // [OUT,G,1] f32
    const float* Bias = (const float*)d_in[4];  // [OUT] f32
    float*       Y    = (float*)d_out;          // [B,S,OUT] f32

    dim3 grid(NDIM / BN, MDIM / BM);            // 128 x 16 blocks
    gql_wmma_kernel<<<grid, 256, 0, stream>>>(X, Wp, Ws, Wb, Bias, Y);
}
